// Quantizer_9706626089635
// MI455X (gfx1250) — matrix operand order repaired and checked by reading, not compiled, not run on hardware
//
#include <hip/hip_runtime.h>

#define D_DIM 256
#define K_DIM 8192
#define N_DIM 16384
#define NT    (K_DIM / 16)          // 512 column tiles
#define NG    (NT / 2)              // 256 double-tile groups
#define TILE_UINTS 2048             // 8 chunks * 32 lanes * 8 dwords per plane tile

typedef __bf16  v16bf __attribute__((ext_vector_type(16)));
typedef float   v8f   __attribute__((ext_vector_type(8)));
typedef int     v4i   __attribute__((vector_size(16)));     // matches builtin param

#if __has_builtin(__builtin_amdgcn_global_load_async_to_lds_b128)
#define HAVE_ASYNC_LDS 1
#endif

__device__ __forceinline__ void copy16_to_lds(const unsigned int* g, unsigned int* l) {
#ifdef HAVE_ASYNC_LDS
    __builtin_amdgcn_global_load_async_to_lds_b128((v4i*)g, (v4i*)l, 0, 0);
#else
    *(uint4*)l = *(const uint4*)g;
#endif
}

__device__ __forceinline__ void wait_async_copies() {
#ifdef HAVE_ASYNC_LDS
#if __has_builtin(__builtin_amdgcn_s_wait_asynccnt)
    __builtin_amdgcn_s_wait_asynccnt(0);
#else
    asm volatile("s_wait_asynccnt 0x0" ::: "memory");
#endif
#endif
}

__device__ __forceinline__ unsigned int pack_bf16(__bf16 lo16, __bf16 hi16) {
    return ((unsigned)__builtin_bit_cast(unsigned short, hi16) << 16)
         |  (unsigned)__builtin_bit_cast(unsigned short, lo16);
}

// ---------------------------------------------------------------------------
// Kernel 1: half squared norms of each codebook column.
// ---------------------------------------------------------------------------
__global__ void vq_norms(const float* __restrict__ e, float* __restrict__ hn) {
    int k = blockIdx.x * blockDim.x + threadIdx.x;
    float s = 0.0f;
    for (int d = 0; d < D_DIM; ++d) {
        float v = e[(size_t)d * K_DIM + k];
        s += v * v;
    }
    hn[k] = 0.5f * s;
}

// ---------------------------------------------------------------------------
// Kernel 1b: one-time conversion of e into bf16 hi/lo planes in B-fragment
// order; each 16-col tile is a contiguous 8KB block per plane.
// uint index u = ((T*8 + cd)*32 + lane)*8 + w  <->
//   col = lane&15, h = lane>>4, d = cd*32 + 16h + 2w, k = T*16 + col
// ---------------------------------------------------------------------------
__global__ void vq_prep(const float* __restrict__ e,
                        unsigned int* __restrict__ eHi,
                        unsigned int* __restrict__ eLo) {
    unsigned u = blockIdx.x * 256 + threadIdx.x;     // 0 .. 2^20-1
    int w    = u & 7;
    int lane = (u >> 3) & 31;
    int cd   = (u >> 8) & 7;
    int T    = u >> 11;
    int col  = lane & 15;
    int h    = lane >> 4;
    int d    = cd * 32 + 16 * h + 2 * w;
    int k    = T * 16 + col;
    float f0 = e[(size_t)d * K_DIM + k];
    float f1 = e[(size_t)(d + 1) * K_DIM + k];
    __bf16 h0 = (__bf16)f0, h1 = (__bf16)f1;
    __bf16 l0 = (__bf16)(f0 - (float)h0);
    __bf16 l1 = (__bf16)(f1 - (float)h1);
    eHi[u] = pack_bf16(h0, h1);
    eLo[u] = pack_bf16(l0, l1);
}

// ---------------------------------------------------------------------------
// Kernel 2 (fast path): fused GEMM + argmax; double-buffered groups of TWO
// 16-col tiles moved with async global->LDS b128 copies (ASYNCcnt).
// ---------------------------------------------------------------------------
__launch_bounds__(256)
__global__ void vq_argmax_pre(const float* __restrict__ x,
                              const unsigned int* __restrict__ eHi,
                              const unsigned int* __restrict__ eLo,
                              const float* __restrict__ hn,
                              int* __restrict__ out_idx) {
    __shared__ alignas(32) unsigned int sHi[2][2 * TILE_UINTS];
    __shared__ alignas(32) unsigned int sLo[2][2 * TILE_UINTS];

    const int t    = threadIdx.x;
    const int lane = t & 31;
    const int wv   = t >> 5;
    const int half = lane >> 4;
    const int l15  = lane & 15;
    const int row  = blockIdx.x * 128 + wv * 16 + l15;

    // Register-resident A fragments (bf16 hi/lo), 16x32 layout.
    v16bf a_hi[8], a_lo[8];
#pragma unroll
    for (int cd = 0; cd < 8; ++cd) {
        v16bf ah, al;
#pragma unroll
        for (int v = 0; v < 8; ++v) {
            int kk = cd * 32 + 2 * v + 16 * half;
            float f0 = x[(size_t)row * D_DIM + kk];
            float f1 = x[(size_t)row * D_DIM + kk + 1];
            __bf16 h0 = (__bf16)f0, h1 = (__bf16)f1;
            ah[2 * v]     = h0;
            ah[2 * v + 1] = h1;
            al[2 * v]     = (__bf16)(f0 - (float)h0);
            al[2 * v + 1] = (__bf16)(f1 - (float)h1);
        }
        a_hi[cd] = ah;
        a_lo[cd] = al;
    }

    float bestv[8];
    int   bestk[8];
#pragma unroll
    for (int j = 0; j < 8; ++j) { bestv[j] = -3.402823e38f; bestk[j] = 0; }

    // Prologue: bring group 0 (tiles 0 and 1) into buffer 0.
#pragma unroll
    for (int s = 0; s < 2; ++s) {
        copy16_to_lds(eHi + (size_t)s * TILE_UINTS + t * 4, &sHi[0][s * TILE_UINTS + t * 4]);
        copy16_to_lds(eLo + (size_t)s * TILE_UINTS + t * 4, &sLo[0][s * TILE_UINTS + t * 4]);
    }
    wait_async_copies();
    __syncthreads();

    int b = 0;
#pragma unroll 1
    for (int G = 0; G < NG; ++G) {
        // Kick off next group's async copies into the other buffer.
        if (G + 1 < NG) {
            const size_t go = (size_t)(G + 1) * (2 * TILE_UINTS);
#pragma unroll
            for (int s = 0; s < 2; ++s) {
                copy16_to_lds(eHi + go + s * TILE_UINTS + t * 4,
                              &sHi[b ^ 1][s * TILE_UINTS + t * 4]);
                copy16_to_lds(eLo + go + s * TILE_UINTS + t * 4,
                              &sLo[b ^ 1][s * TILE_UINTS + t * 4]);
            }
        }

        // Two subtiles computed sequentially: 48 WMMAs between barriers.
#pragma unroll
        for (int s = 0; s < 2; ++s) {
            const unsigned int* baseH = &sHi[b][s * TILE_UINTS];
            const unsigned int* baseL = &sLo[b][s * TILE_UINTS];
            v8f acc0 = {}, acc1 = {};
            v16bf bh = *(const v16bf*)&baseH[lane * 8];
            v16bf bl = *(const v16bf*)&baseL[lane * 8];
#pragma unroll
            for (int cd = 0; cd < 8; ++cd) {
                acc0 = __builtin_amdgcn_wmma_f32_16x16x32_bf16(
                           false, a_hi[cd], false, bh, (short)0, acc0, false, false);
                acc1 = __builtin_amdgcn_wmma_f32_16x16x32_bf16(
                           false, a_lo[cd], false, bh, (short)0, acc1, false, false);
                acc1 = __builtin_amdgcn_wmma_f32_16x16x32_bf16(
                           false, a_hi[cd], false, bl, (short)0, acc1, false, false);
                if (cd < 7) {
                    bh = *(const v16bf*)&baseH[((cd + 1) * 32 + lane) * 8];
                    bl = *(const v16bf*)&baseL[((cd + 1) * 32 + lane) * 8];
                }
            }

            int   col = (2 * G + s) * 16 + l15;
            float h   = hn[col];
#pragma unroll
            for (int j = 0; j < 8; ++j) {
                float sc = (acc0[j] + acc1[j]) - h;
                if (sc > bestv[j]) { bestv[j] = sc; bestk[j] = col; }
            }
        }

        wait_async_copies();
        __syncthreads();
        b ^= 1;
    }

#pragma unroll
    for (int j = 0; j < 8; ++j) {
        float v  = bestv[j];
        int   kk = bestk[j];
#pragma unroll
        for (int m = 8; m >= 1; m >>= 1) {
            float ov = __shfl_xor(v, m, 32);
            int   ok = __shfl_xor(kk, m, 32);
            if (ov > v || (ov == v && ok < kk)) { v = ov; kk = ok; }
        }
        if (l15 == 0) {
            int r = blockIdx.x * 128 + wv * 16 + 8 * half + j;
            out_idx[r] = kk;
        }
    }
}

// ---------------------------------------------------------------------------
// Kernel 2 (fallback path, small workspace): in-kernel staging.
// ---------------------------------------------------------------------------
__launch_bounds__(256)
__global__ void vq_argmax(const float* __restrict__ x, const float* __restrict__ e,
                          const float* __restrict__ hn, int* __restrict__ out_idx) {
    __shared__ alignas(32) unsigned int sBhi[8 * 32 * 8];
    __shared__ alignas(32) unsigned int sBlo[8 * 32 * 8];

    const int t    = threadIdx.x;
    const int lane = t & 31;
    const int wv   = t >> 5;
    const int half = lane >> 4;
    const int l15  = lane & 15;
    const int row  = blockIdx.x * 128 + wv * 16 + l15;

    v16bf a_hi[8], a_lo[8];
#pragma unroll
    for (int cd = 0; cd < 8; ++cd) {
        v16bf ah, al;
#pragma unroll
        for (int v = 0; v < 8; ++v) {
            int kk = cd * 32 + 2 * v + 16 * half;
            float f0 = x[(size_t)row * D_DIM + kk];
            float f1 = x[(size_t)row * D_DIM + kk + 1];
            __bf16 h0 = (__bf16)f0, h1 = (__bf16)f1;
            ah[2 * v]     = h0;
            ah[2 * v + 1] = h1;
            al[2 * v]     = (__bf16)(f0 - (float)h0);
            al[2 * v + 1] = (__bf16)(f1 - (float)h1);
        }
        a_hi[cd] = ah;
        a_lo[cd] = al;
    }

    float bestv[8];
    int   bestk[8];
#pragma unroll
    for (int j = 0; j < 8; ++j) { bestv[j] = -3.402823e38f; bestk[j] = 0; }

    for (int k0 = 0; k0 < K_DIM; k0 += 16) {
        __syncthreads();
#pragma unroll
        for (int i = 0; i < 8; ++i) {
            int q   = t + 256 * i;
            int col = q & 15;
            int d   = (q >> 4) * 2;
            const float* ep = e + (size_t)d * K_DIM + k0 + col;
            float f0 = ep[0];
            float f1 = ep[K_DIM];
            __bf16 h0 = (__bf16)f0, h1 = (__bf16)f1;
            __bf16 l0 = (__bf16)(f0 - (float)h0);
            __bf16 l1 = (__bf16)(f1 - (float)h1);
            int cd = d >> 5;
            int dd = d & 31;
            int la = col + 16 * (dd >> 4);
            int wi = (cd * 32 + la) * 8 + ((dd & 15) >> 1);
            sBhi[wi] = pack_bf16(h0, h1);
            sBlo[wi] = pack_bf16(l0, l1);
        }
        __syncthreads();

        v8f acc0 = {}, acc1 = {};
#pragma unroll
        for (int cd = 0; cd < 8; ++cd) {
            v16bf bh = *(const v16bf*)&sBhi[(cd * 32 + lane) * 8];
            v16bf bl = *(const v16bf*)&sBlo[(cd * 32 + lane) * 8];
            acc0 = __builtin_amdgcn_wmma_f32_16x16x32_bf16(
                       false, a_hi[cd], false, bh, (short)0, acc0, false, false);
            acc1 = __builtin_amdgcn_wmma_f32_16x16x32_bf16(
                       false, a_lo[cd], false, bh, (short)0, acc1, false, false);
            acc1 = __builtin_amdgcn_wmma_f32_16x16x32_bf16(
                       false, a_hi[cd], false, bl, (short)0, acc1, false, false);
        }

        int   col = k0 + l15;
        float h   = hn[col];
#pragma unroll
        for (int j = 0; j < 8; ++j) {
            float s = (acc0[j] + acc1[j]) - h;
            if (s > bestv[j]) { bestv[j] = s; bestk[j] = col; }
        }
    }

#pragma unroll
    for (int j = 0; j < 8; ++j) {
        float v  = bestv[j];
        int   kk = bestk[j];
#pragma unroll
        for (int m = 8; m >= 1; m >>= 1) {
            float ov = __shfl_xor(v, m, 32);
            int   ok = __shfl_xor(kk, m, 32);
            if (ov > v || (ov == v && ok < kk)) { v = ov; kk = ok; }
        }
        if (l15 == 0) {
            int r = blockIdx.x * 128 + wv * 16 + 8 * half + j;
            out_idx[r] = kk;
        }
    }
}

// ---------------------------------------------------------------------------
// Kernel 3: gather quantized rows, straight-through output, SSE + histogram.
// ---------------------------------------------------------------------------
__global__ void vq_gather(const float* __restrict__ x, const float* __restrict__ e,
                          const int* __restrict__ idx, float* __restrict__ out,
                          float* __restrict__ counts, float* __restrict__ sse) {
    __shared__ float red[256];
    const int n = blockIdx.x;
    const int t = threadIdx.x;
    const int k = idx[n];
    float xv  = x[(size_t)n * D_DIM + t];
    float q   = e[(size_t)t * K_DIM + k];
    float qst = xv + (q - xv);
    out[(size_t)n * D_DIM + t] = qst;
    float d = q - xv;
    red[t] = d * d;
    __syncthreads();
    for (int s = 128; s > 0; s >>= 1) {
        if (t < s) red[t] += red[t + s];
        __syncthreads();
    }
    if (t == 0) {
        atomicAdd(sse, red[0]);
        atomicAdd(&counts[k], 1.0f);
        out[(size_t)N_DIM * D_DIM + 1 + n] = (float)k;
    }
}

// ---------------------------------------------------------------------------
// Kernel 4: loss, avg_probs, perplexity.
// ---------------------------------------------------------------------------
__global__ void vq_finalize(const float* __restrict__ counts,
                            const float* __restrict__ sse,
                            float* __restrict__ out) {
    __shared__ float red[256];
    const int t = threadIdx.x;
    float hsum = 0.0f;
    for (int k = t; k < K_DIM; k += 256) {
        float p = counts[k] * (1.0f / (float)N_DIM);
        out[(size_t)N_DIM * D_DIM + 1 + N_DIM + k] = p;
        hsum += p * logf(p + 1e-20f);
    }
    red[t] = hsum;
    __syncthreads();
    for (int s = 128; s > 0; s >>= 1) {
        if (t < s) red[t] += red[t + s];
        __syncthreads();
    }
    if (t == 0) {
        out[(size_t)N_DIM * D_DIM] =
            1.25f * sse[0] / (float)((size_t)N_DIM * D_DIM);
        out[(size_t)N_DIM * D_DIM + 1 + N_DIM + K_DIM] = expf(-red[0]);
    }
}

// ---------------------------------------------------------------------------
// d_out (floats): [quantized_st N*D][loss 1][enc_idx N][avg_probs K][ppl 1]
// ws     (floats): [counts K][sse 1][hn K] [idx: N ints] [pad] [eHi 4MB][eLo 4MB]
// ---------------------------------------------------------------------------
extern "C" void kernel_launch(void* const* d_in, const int* in_sizes, int n_in,
                              void* d_out, int out_size, void* d_ws, size_t ws_size,
                              hipStream_t stream) {
    const float* x = (const float*)d_in[0];
    const float* e = (const float*)d_in[1];
    float* out    = (float*)d_out;
    float* ws     = (float*)d_ws;
    float* counts = ws;
    float* sse    = ws + K_DIM;
    float* hn     = ws + K_DIM + 1;
    int*   idx    = (int*)(ws + 2 * K_DIM + 2);

    const size_t planeUints = (size_t)NT * TILE_UINTS;        // 1,048,576
    const size_t eHiOffF    = 2 * K_DIM + 2 + N_DIM + 2;      // 16B-aligned float offset
    unsigned int* eHi = (unsigned int*)(ws + eHiOffF);
    unsigned int* eLo = eHi + planeUints;
    const size_t needBytes = eHiOffF * sizeof(float) + 2 * planeUints * sizeof(unsigned int);

    (void)hipMemsetAsync(counts, 0, (K_DIM + 1) * sizeof(float), stream);
    vq_norms<<<K_DIM / 256, 256, 0, stream>>>(e, hn);

    if (ws_size >= needBytes) {
        vq_prep      <<<(int)(planeUints / 256), 256, 0, stream>>>(e, eHi, eLo);
        vq_argmax_pre<<<N_DIM / 128, 256, 0, stream>>>(x, eHi, eLo, hn, idx);
    } else {
        vq_argmax    <<<N_DIM / 128, 256, 0, stream>>>(x, e, hn, idx);
    }

    vq_gather  <<<N_DIM, 256, 0, stream>>>(x, e, idx, out, counts, sse);
    vq_finalize<<<1,     256, 0, stream>>>(counts, sse, out);
}
